// ACmix1d_48979807044039
// MI455X (gfx1250) — matrix operand order repaired and checked by reading, not compiled, not run on hardware
//
#include <hip/hip_runtime.h>

typedef __attribute__((ext_vector_type(16))) _Float16 v16h;
typedef __attribute__((ext_vector_type(8)))  _Float16 v8h;
typedef __attribute__((ext_vector_type(8)))  float    v8f;

constexpr int CIN  = 256;
constexpr int COUT = 256;
constexpr int WW   = 8192;
constexpr int BB   = 4;
constexpr int HD   = 64;
constexpr int HH   = 4;
constexpr int KATT = 7;
constexpr int PADA = 3;

// ---------------------------------------------------------------------------
// CDNA5 async global->LDS copy (ASYNCcnt-tracked DMA, no VGPR round trip).
// lds_off: byte offset inside the workgroup LDS allocation (VDST operand).
// ---------------------------------------------------------------------------
__device__ __forceinline__ void async_ld_b128(unsigned lds_off, const void* gsrc) {
  asm volatile("global_load_async_to_lds_b128 %0, %1, off"
               :: "v"(lds_off), "v"(gsrc) : "memory");
}
__device__ __forceinline__ void wait_async_le1() {
  asm volatile("s_wait_asynccnt 0x1" ::: "memory");
}
__device__ __forceinline__ void wait_async_0() {
  asm volatile("s_wait_asynccnt 0x0" ::: "memory");
}

// ---------------------------------------------------------------------------
// Kernel 0a: convert the three 256x256 f32 weight matrices to f16
// ---------------------------------------------------------------------------
__global__ void wcvt_kernel(const float* __restrict__ w1,
                            const float* __restrict__ w2,
                            const float* __restrict__ w3,
                            _Float16* __restrict__ wf) {
  int idx = blockIdx.x * blockDim.x + threadIdx.x;
  if (idx >= 3 * COUT * CIN) return;
  int mtx = idx >> 16;
  int r   = idx & 65535;
  const float* s = (mtx == 0) ? w1 : (mtx == 1) ? w2 : w3;
  wf[idx] = (_Float16)s[r];
}

// ---------------------------------------------------------------------------
// Kernel 0b: x [b][c][w] f32 -> xT [b][w][c] f16 (tiled LDS transpose).
// Channel-major xT makes the GEMM's LDS [n][k] tile row-contiguous so it can
// be staged with async b128 DMA, and is reused by nothing else.
// ---------------------------------------------------------------------------
__global__ __launch_bounds__(256)
void xcvt_kernel(const float* __restrict__ x, _Float16* __restrict__ xT) {
  __shared__ _Float16 tile[32][33];
  const int bw = blockIdx.x;            // w tile (W/32 = 256)
  const int bc = blockIdx.y & 7;        // c tile (C/32 = 8)
  const int b  = blockIdx.y >> 3;
  const int tx = threadIdx.x & 31;
  const int ty = threadIdx.x >> 5;      // 8 rows per pass
  const int w0 = bw * 32, c0 = bc * 32;
  #pragma unroll
  for (int i = 0; i < 4; ++i) {
    int c = ty + i * 8;
    tile[c][tx] = (_Float16)x[((size_t)b * CIN + c0 + c) * WW + w0 + tx];
  }
  __syncthreads();
  #pragma unroll
  for (int i = 0; i < 4; ++i) {
    int wr = ty + i * 8;
    xT[((size_t)b * WW + w0 + wr) * CIN + c0 + tx] = tile[tx][wr];
  }
}

// ---------------------------------------------------------------------------
// Kernel 1: q/k/v GEMM via v_wmma_f32_16x16x32_f16 with double-buffered
// async global->LDS staging of the xT tile.
// Block: 256 threads (8 waves). Computes [COUT=256] x [64 w] of one of
// q/k/v (blockIdx.y). Wave wv owns M-range [32wv, 32wv+32).
// LDS tile layout [n=64][k=32] f16 (row-contiguous in xT) -> B fragments are
// one 32B contiguous read; outputs stored channel-major [b][w][c] as v8h.
// ---------------------------------------------------------------------------
__global__ __launch_bounds__(256)
void qkv_gemm_kernel(const _Float16* __restrict__ xT,  // [B][W][256] f16
                     const _Float16* __restrict__ wf,  // [3][256][256] f16
                     const float* __restrict__ b1,
                     const float* __restrict__ b2,
                     const float* __restrict__ b3,
                     _Float16* __restrict__ qkv)       // [3][B][W][256] f16
{
  __shared__ __align__(32) _Float16 xs[2][64 * 32];    // 2 x 4KB double buffer

  const int mtx  = blockIdx.y;
  const int bb   = blockIdx.x >> 7;           // batch
  const int w0   = (blockIdx.x & 127) * 64;   // N offset inside batch
  const int tid  = threadIdx.x;
  const int lane = tid & 31;
  const int wv   = tid >> 5;
  const int m0   = wv * 32;
  const int l16  = lane & 15;
  const int lhalf = lane >> 4;

  const _Float16* wmat = wf + (size_t)mtx * COUT * CIN;
  const float* bias = (mtx == 0) ? b1 : (mtx == 1) ? b2 : b3;

  // async staging: 256 threads x 16B = 4KB tile; thread t copies row nn4,
  // half-offset ko8 -> LDS byte t*16 (contiguous on both sides).
  const _Float16* xtb = xT + ((size_t)bb * WW + w0) * CIN;
  const unsigned ldsbase = (unsigned)(size_t)(&xs[0][0]);
  const int nn4 = tid >> 2;
  const int ko8 = (tid & 3) * 8;

  v8f acc[2][4];
  #pragma unroll
  for (int i = 0; i < 2; ++i)
    #pragma unroll
    for (int j = 0; j < 4; ++j) acc[i][j] = v8f{};

  // prologue: issue DMA for K-step 0 into buffer 0
  async_ld_b128(ldsbase + (unsigned)tid * 16,
                xtb + (size_t)nn4 * CIN + ko8);

  for (int ks = 0; ks < 8; ++ks) {
    const int cur = ks & 1;
    const int c0  = ks * 32;

    if (ks < 7) {  // overlap next tile's DMA with this step's WMMAs
      async_ld_b128(ldsbase + (unsigned)(cur ^ 1) * 4096 + (unsigned)tid * 16,
                    xtb + (size_t)nn4 * CIN + (ks + 1) * 32 + ko8);
      wait_async_le1();   // oldest (current buffer) done; newest may fly
    } else {
      wait_async_0();
    }
    __syncthreads();      // publish DMA-written LDS to all waves

    // A fragments: contiguous k order (k = 16*lhalf + s), same as B
    // lanes 0-15 : M=lane,   halfs0-15=K0-15
    // lanes 16-31: M=lane-16, halfs0-15=K16-31
    v16h afrag[2];
    #pragma unroll
    for (int mt = 0; mt < 2; ++mt) {
      int m = m0 + mt * 16 + l16;
      const _Float16* wr = wmat + (size_t)m * CIN + c0 + lhalf * 16;
      v8h lo = *(const v8h*)(wr);
      v8h hi = *(const v8h*)(wr + 8);
      #pragma unroll
      for (int i = 0; i < 8; ++i) { afrag[mt][i] = lo[i]; afrag[mt][8 + i] = hi[i]; }
    }

    // B fragments: lane n = l16; lanes 0-15 hold K0-15, lanes 16-31 K16-31
    #pragma unroll
    for (int nt = 0; nt < 4; ++nt) {
      int nl = nt * 16 + l16;
      v16h bfrag = *(const v16h*)(xs[cur] + nl * 32 + lhalf * 16);
      #pragma unroll
      for (int mt = 0; mt < 2; ++mt) {
        acc[mt][nt] = __builtin_amdgcn_wmma_f32_16x16x32_f16(
            false, afrag[mt], false, bfrag, (short)0, acc[mt][nt], false, false);
      }
    }
    __syncthreads();      // all waves done reading before buffer is re-filled
  }

  // Epilogue: C/D layout lane n=l16, VGPR r -> m = r + 8*lhalf.
  // Channel-major store: the 8 accumulator rows are 8 consecutive channels
  // -> one packed v8h 16B store per tile.
  _Float16* outb = qkv + ((size_t)mtx * BB + bb) * WW * COUT + (size_t)w0 * COUT;
  #pragma unroll
  for (int mt = 0; mt < 2; ++mt) {
    #pragma unroll
    for (int nt = 0; nt < 4; ++nt) {
      int n = nt * 16 + l16;
      int mbase = m0 + mt * 16 + lhalf * 8;
      v8h pk;
      #pragma unroll
      for (int r = 0; r < 8; ++r)
        pk[r] = (_Float16)(acc[mt][nt][r] + bias[mbase + r]);
      *(v8h*)(outb + (size_t)n * COUT + mbase) = pk;
    }
  }
}

// ---------------------------------------------------------------------------
// Kernel 2: fused windowed attention (K=7, reflect pad) + conv branch.
// qkvT is channel-major so every access is a contiguous b128 load.
// pe term factored exactly: att[k] = q.k_j + dloc[k] * (q.wp)  (bp cancels).
// ---------------------------------------------------------------------------
__global__ __launch_bounds__(128)
void attn_conv_kernel(const _Float16* __restrict__ qT,  // [B][W][256]
                      const _Float16* __restrict__ kT,
                      const _Float16* __restrict__ vT,
                      const float* __restrict__ wp,
                      const float* __restrict__ fcw,
                      const float* __restrict__ depw, const float* __restrict__ depb,
                      const float* __restrict__ rate1, const float* __restrict__ rate2,
                      float* __restrict__ out)
{
  __shared__ float fconv[HD * 128];     // 32KB, lane-major (conflict-free)
  const int b   = blockIdx.y;
  const int tid = threadIdx.x;
  const int w   = blockIdx.x * 128 + tid;

  int   jk[KATT];
  float dloc[KATT];
  const float locScale = 2.0f / (float)(WW - 1);
  const float loc_w = -1.0f + locScale * (float)w;
  #pragma unroll
  for (int kk = 0; kk < KATT; ++kk) {
    int j = w + kk - PADA;
    if (j < 0)   j = -j;
    if (j >= WW) j = 2 * WW - 2 - j;
    jk[kk]   = j;
    dloc[kk] = loc_w - (-1.0f + locScale * (float)j);
  }

  const size_t rowW = (size_t)b * WW;
  const _Float16* qrow = qT + (rowW + w) * COUT;
  const _Float16* krow = kT + (rowW + w) * COUT;
  const _Float16* vrow = vT + (rowW + w) * COUT;

  float fq[HH], fk[HH], fv[HH];
  #pragma unroll
  for (int h = 0; h < HH; ++h) { fq[h] = fcw[h]; fk[h] = fcw[HH + h]; fv[h] = fcw[2 * HH + h]; }

  // conv-branch feature f_conv[b, dg, w], dg = head-dim index
  #pragma unroll
  for (int d8 = 0; d8 < 8; ++d8) {
    float a8[8];
    #pragma unroll
    for (int i = 0; i < 8; ++i) a8[i] = 0.f;
    #pragma unroll
    for (int h = 0; h < HH; ++h) {
      v8h q8 = *(const v8h*)(qrow + h * HD + d8 * 8);
      v8h k8 = *(const v8h*)(krow + h * HD + d8 * 8);
      v8h v8_ = *(const v8h*)(vrow + h * HD + d8 * 8);
      #pragma unroll
      for (int i = 0; i < 8; ++i)
        a8[i] += fq[h] * (float)q8[i] + fk[h] * (float)k8[i] + fv[h] * (float)v8_[i];
    }
    #pragma unroll
    for (int i = 0; i < 8; ++i) fconv[(d8 * 8 + i) * 128 + tid] = a8[i];
  }
  __syncthreads();

  const float r1 = rate1[0], r2 = rate2[0];
  const float scaling = 0.125f;   // HD^-0.5

  for (int h = 0; h < HH; ++h) {
    float att[KATT];
    #pragma unroll
    for (int kk = 0; kk < KATT; ++kk) att[kk] = 0.f;
    float S = 0.f;   // q . wp  (positional factor)

    const _Float16* qh = qrow + h * HD;
    #pragma unroll
    for (int d8 = 0; d8 < 8; ++d8) {
      v8h q8 = *(const v8h*)(qh + d8 * 8);
      float qf[8];
      #pragma unroll
      for (int i = 0; i < 8; ++i) {
        qf[i] = (float)q8[i] * scaling;
        S += qf[i] * wp[d8 * 8 + i];
      }
      #pragma unroll
      for (int kk = 0; kk < KATT; ++kk) {
        v8h k8 = *(const v8h*)(kT + (rowW + jk[kk]) * COUT + h * HD + d8 * 8);
        #pragma unroll
        for (int i = 0; i < 8; ++i) att[kk] += qf[i] * (float)k8[i];
      }
    }
    #pragma unroll
    for (int kk = 0; kk < KATT; ++kk) att[kk] += S * dloc[kk];

    // softmax over the 7-wide window
    float mx = att[0];
    #pragma unroll
    for (int kk = 1; kk < KATT; ++kk) mx = fmaxf(mx, att[kk]);
    float sum = 0.f;
    #pragma unroll
    for (int kk = 0; kk < KATT; ++kk) { att[kk] = __expf(att[kk] - mx); sum += att[kk]; }
    float inv = 1.0f / sum;
    #pragma unroll
    for (int kk = 0; kk < KATT; ++kk) att[kk] *= inv;

    // out[b, o=h*64+d, w]; dep_conv group g = o>>2 = h*16 + (d>>2)
    #pragma unroll
    for (int d8 = 0; d8 < 8; ++d8) {
      float acc8[8];
      #pragma unroll
      for (int i = 0; i < 8; ++i) acc8[i] = 0.f;
      #pragma unroll
      for (int kk = 0; kk < KATT; ++kk) {
        v8h v8_ = *(const v8h*)(vT + (rowW + jk[kk]) * COUT + h * HD + d8 * 8);
        #pragma unroll
        for (int i = 0; i < 8; ++i) acc8[i] += att[kk] * (float)v8_[i];
      }
      #pragma unroll
      for (int i = 0; i < 8; ++i) {
        int d = d8 * 8 + i;
        int o = h * HD + d;
        int g = (h * 16) + (d >> 2);
        float res = r1 * acc8[i] + r2 * (depw[o] * fconv[g * 128 + tid] + depb[o]);
        out[((size_t)b * COUT + o) * WW + w] = res;
      }
    }
  }
}

// ---------------------------------------------------------------------------
extern "C" void kernel_launch(void* const* d_in, const int* in_sizes, int n_in,
                              void* d_out, int out_size, void* d_ws, size_t ws_size,
                              hipStream_t stream) {
  const float* x    = (const float*)d_in[0];
  const float* w1   = (const float*)d_in[1];
  const float* b1   = (const float*)d_in[2];
  const float* w2   = (const float*)d_in[3];
  const float* b2   = (const float*)d_in[4];
  const float* w3   = (const float*)d_in[5];
  const float* b3   = (const float*)d_in[6];
  const float* wp   = (const float*)d_in[7];
  // d_in[8] = bp: cancels exactly in pe - unfold(pe); unused
  const float* fcw  = (const float*)d_in[9];
  const float* depw = (const float*)d_in[10];
  const float* depb = (const float*)d_in[11];
  const float* r1   = (const float*)d_in[12];
  const float* r2   = (const float*)d_in[13];
  float* out = (float*)d_out;

  // workspace: [3*256*256 f16 W][B*W*256 f16 xT][3*B*W*256 f16 qkvT] ~ 65MB
  // (entire working set is resident in the 192MB L2)
  char* wsb = (char*)d_ws;
  _Float16* wf  = (_Float16*)wsb;
  _Float16* xT  = (_Float16*)(wsb + (size_t)3 * COUT * CIN * sizeof(_Float16));
  _Float16* qkv = (_Float16*)(wsb + (size_t)3 * COUT * CIN * sizeof(_Float16)
                                  + (size_t)BB * WW * CIN * sizeof(_Float16));

  wcvt_kernel<<<dim3((3 * COUT * CIN + 255) / 256), dim3(256), 0, stream>>>(w1, w2, w3, wf);
  xcvt_kernel<<<dim3(WW / 32, (CIN / 32) * BB), dim3(256), 0, stream>>>(x, xT);

  qkv_gemm_kernel<<<dim3(512, 3), dim3(256), 0, stream>>>(xT, wf, b1, b2, b3, qkv);

  const _Float16* qb = qkv;
  const _Float16* kb = qkv + (size_t)BB * WW * COUT;
  const _Float16* vb = qkv + (size_t)2 * BB * WW * COUT;
  attn_conv_kernel<<<dim3(WW / 128, BB), dim3(128), 0, stream>>>(
      qb, kb, vb, wp, fcw, depw, depb, r1, r2, out);
}
